// CovaMLoss_22926535426716
// MI455X (gfx1250) — matrix operand order repaired and checked by reading, not compiled, not run on hardware
//
#include <hip/hip_runtime.h>
#include <hip/hip_bf16.h>

// CDNA5 / gfx1250: wave32, WMMA 16x16x32 bf16 -> f32
typedef __attribute__((ext_vector_type(16))) __bf16 v16bf;
typedef __attribute__((ext_vector_type(8)))  __bf16 v8bf;
typedef __attribute__((ext_vector_type(8)))  float  v8f;

#define CCH   32          // channels
#define KCLS  16          // classes
#define NPIX  4096        // H*W
#define COLS  256         // columns per block
#define SROW  40          // padded LDS row stride (halfs) for S: 80B, 16B aligned

union V16U { v16bf v; v8bf h[2]; };

__device__ __forceinline__ unsigned short f2bf(float f) {
    unsigned int u = __float_as_uint(f);
    unsigned int r = u + 0x7fffu + ((u >> 16) & 1u);   // round-to-nearest-even
    return (unsigned short)(r >> 16);
}
__device__ __forceinline__ float bf2f(unsigned short h) {
    return __uint_as_float(((unsigned int)h) << 16);
}
// Swap lane halves of a wave32: ds_swizzle SWAPX16 (xor=0x10, or=0, and=0x1f)
__device__ __forceinline__ float swap16(float x) {
    return __int_as_float(__builtin_amdgcn_ds_swizzle(__float_as_int(x), 0x401f));
}

// ---- kernel 1: per (b,c) row inverse L2 norm over 4096 pixels ----
__global__ __launch_bounds__(256) void cova_norm_kernel(const float* __restrict__ inp,
                                                        float* __restrict__ invnorm) {
    __shared__ float red[8];
    const int row = blockIdx.x;                 // b*CCH + c
    const float* p = inp + (size_t)row * NPIX;
    float s = 0.f;
    for (int i = threadIdx.x; i < NPIX; i += 256) { float v = p[i]; s += v * v; }
    for (int off = 16; off > 0; off >>= 1) s += __shfl_down(s, off, 32);
    if ((threadIdx.x & 31) == 0) red[threadIdx.x >> 5] = s;
    __syncthreads();
    if (threadIdx.x == 0) {
        float tot = 0.f;
        #pragma unroll
        for (int i = 0; i < 8; ++i) tot += red[i];
        invnorm[row] = rsqrtf(tot);
    }
}

// ---- kernel 2: sim[b,k,n] = q(:,n)^T S_k q(:,n) via WMMA ----
__global__ __launch_bounds__(256) void cova_sim_kernel(const float* __restrict__ inp,
                                                       const float* __restrict__ covas,
                                                       const float* __restrict__ invnorm,
                                                       float* __restrict__ out) {
    __shared__ unsigned short sS[KCLS * CCH * SROW];   // S_k, bf16, padded rows (40 KB)
    __shared__ unsigned short sQ[COLS * CCH];          // q tile, column-major bf16 (16 KB)

    const int b  = blockIdx.y;
    const int n0 = blockIdx.x * COLS;
    const int t  = threadIdx.x;

    // Stage all K covariance matrices (coalesced global read, bf16 to LDS)
    for (int i = t; i < KCLS * CCH * CCH; i += 256) {
        const int k   = i >> 10;
        const int row = (i >> 5) & 31;
        const int col = i & 31;
        sS[(k * CCH + row) * SROW + col] = f2bf(covas[i]);
    }
    // Stage normalized q tile, column-major: sQ[col*32 + c]
    {
        const int col = t;
        const float* src = inp + (size_t)b * (CCH * NPIX) + n0 + col;
        #pragma unroll 8
        for (int c = 0; c < CCH; ++c) {
            const float inv = invnorm[b * CCH + c];     // uniform -> scalar load
            sQ[col * CCH + c] = f2bf(src[(size_t)c * NPIX] * inv);
        }
    }
    __syncthreads();

    const int wave    = t >> 5;
    const int lane    = t & 31;
    const int half    = lane >> 4;      // 0: lanes 0-15, 1: lanes 16-31
    const int l15     = lane & 15;
    const int colbase = wave * 32;      // this wave: 32 columns = two 16-wide N tiles

    // B fragments (K x N = 32 x 16, bf16): lane<16 holds K=0..15 of col=lane,
    // lane>=16 holds K=16..31 of col=lane-16 -> contiguous 32B per lane (column-major LDS).
    v16bf bfrag[2];
    float qd[2][16];                    // q values matching the C/D row layout
    #pragma unroll
    for (int nt = 0; nt < 2; ++nt) {
        const int col = colbase + nt * 16 + l15;
        const unsigned short* qcol = &sQ[col * CCH];
        bfrag[nt] = *reinterpret_cast<const v16bf*>(qcol + half * 16);
        const int cb = half * 8;        // D rows: lane<16 -> M=j, lane>=16 -> M=j+8
        #pragma unroll
        for (int j = 0; j < 8; ++j) {
            qd[nt][j]     = bf2f(qcol[cb + j]);        // c in [cb, cb+8)      (A rows 0-15)
            qd[nt][8 + j] = bf2f(qcol[cb + 16 + j]);   // c in [cb+16, cb+24)  (A rows 16-31)
        }
    }

    // A fragment addressing (contiguous K order, same as the B fragments):
    // lane<16: row M=lane, halfs K=0..15 ; lane>=16: row M=lane-16, K=16..31
    const int arow = l15;
    const int acb  = half * 16;
    float* outb = out + (size_t)b * (KCLS * NPIX) + n0 + colbase + lane;

    for (int k = 0; k < KCLS; ++k) {
        V16U a0, a1;
        const unsigned short* s0 = &sS[(k * CCH + arow) * SROW + acb];
        const unsigned short* s1 = &sS[(k * CCH + 16 + arow) * SROW + acb];
        a0.h[0] = *reinterpret_cast<const v8bf*>(s0);
        a0.h[1] = *reinterpret_cast<const v8bf*>(s0 + 8);
        a1.h[0] = *reinterpret_cast<const v8bf*>(s1);
        a1.h[1] = *reinterpret_cast<const v8bf*>(s1 + 8);

        float full[2];
        #pragma unroll
        for (int nt = 0; nt < 2; ++nt) {
            v8f acc0 = {};
            v8f acc1 = {};
            // T(0:16, tile) and T(16:32, tile): full C=32 contraction in one K pass each
            acc0 = __builtin_amdgcn_wmma_f32_16x16x32_bf16(false, a0.v, false, bfrag[nt],
                                                           (short)0, acc0, false, false);
            acc1 = __builtin_amdgcn_wmma_f32_16x16x32_bf16(false, a1.v, false, bfrag[nt],
                                                           (short)0, acc1, false, false);
            // diag dot: sim(col) = sum_c q[c,col] * T[c,col], split across lane halves
            float p = acc0[0] * qd[nt][0];
            #pragma unroll
            for (int j = 1; j < 8; ++j) p += acc0[j] * qd[nt][j];
            #pragma unroll
            for (int j = 0; j < 8; ++j) p += acc1[j] * qd[nt][8 + j];
            full[nt] = p + swap16(p);        // both halves now hold the column sum
        }
        // Lanes 0-15 carry tile 0 (cols colbase+0..15), lanes 16-31 carry tile 1
        // (cols colbase+16..31): one unpredicated, fully coalesced 128B store.
        outb[(size_t)k * NPIX] = half ? full[1] : full[0];
    }
}

extern "C" void kernel_launch(void* const* d_in, const int* in_sizes, int n_in,
                              void* d_out, int out_size, void* d_ws, size_t ws_size,
                              hipStream_t stream) {
    const float* inp    = (const float*)d_in[0];   // [64, 32, 64, 64] f32
    const float* covas  = (const float*)d_in[1];   // [16, 32, 32]     f32
    float* invnorm      = (float*)d_ws;            // [64*32] f32 scratch
    float* out          = (float*)d_out;           // [64, 16*4096]    f32

    cova_norm_kernel<<<dim3(64 * CCH), 256, 0, stream>>>(inp, invnorm);
    cova_sim_kernel<<<dim3(NPIX / COLS, 64), 256, 0, stream>>>(inp, covas, invnorm, out);
}
